// LSTMLSTM_35991825940830
// MI455X (gfx1250) — matrix operand order repaired and checked by reading, not compiled, not run on hardware
//
#include <hip/hip_runtime.h>

// ---------------------------------------------------------------------------
// 2-layer LSTM (B=256, T=512, I=64, H=512) + FC(512->1), fused persistent
// kernel for gfx1250 using v_wmma_f32_16x16x32_bf16.
//
// 16 workgroups x 1024 threads (32 waves). Each WG owns a 16-row batch tile
// for the whole T=512 recurrence of both layers. Wave w owns hidden columns
// [16w,16w+16): it computes the 4 gate tiles (i,f,g,o) for those columns via
// WMMA over K = [h_prev | input], then performs the cell update in registers.
// h tiles cross waves through LDS (bf16).
//
// Register discipline (32 resident waves => ~128 VGPR/wave budget): exactly
// 4 live B fragments + 1 A fragment. The 4 gate fragments are loaded as one
// clause through a SINGLE address register pair (gate-row strides fit the
// 24-bit global-load immediate), then consumed by 4 WMMAs with staged
// partial loadcnt waits. k-loops are rolled (#pragma unroll 1) and
// segmented by A-source: no aperture selects, no scheduler-induced spills.
// Gate nonlinearities use hardware v_exp/v_rcp transcendentals (the OCML
// tanhf polynomial doubled the VALU count in round 3).
// ---------------------------------------------------------------------------

typedef __attribute__((ext_vector_type(16))) __bf16       v16bf;
typedef __attribute__((ext_vector_type(8)))  float        v8f;
typedef __attribute__((ext_vector_type(4)))  unsigned int v4u;

union Frag16 {
    v16bf v;
    v4u   u[2];
};

__device__ inline v16bf load_frag_2x16B(const __bf16* p0, const __bf16* p1) {
    Frag16 f;
    f.u[0] = *(const v4u*)p0;   // 8 bf16 -> elements 0..7  (K .. K+7)
    f.u[1] = *(const v4u*)p1;   // 8 bf16 -> elements 8..15 (K+16 .. K+23)
    return f.v;
}

__device__ inline v16bf load_frag_32B(const __bf16* p) {
    Frag16 f;
    f.u[0] = *(const v4u*)p;        // 8 bf16 along K (K .. K+7), then K+16 .. K+23 (B fragment)
    f.u[1] = *(const v4u*)(p + 16);
    return f.v;
}

// Hardware-transcendental gate functions: v_exp_f32 + v_rcp_f32, which
// co-execute with the XDL/WMMA pipe (trans ops).
__device__ inline float sigm_f(float x) {
    return __builtin_amdgcn_rcpf(1.0f + __expf(-x));
}
__device__ inline float tanh_f(float x) {
    return 2.0f * __builtin_amdgcn_rcpf(1.0f + __expf(-2.0f * x)) - 1.0f;
}

#define BSZ  256
#define TT   512
#define IN   64
#define HH   512
#define K1   576     // 512 (h1) + 64 (x)
#define K2   1024    // 512 (h2) + 512 (h1)
#define G1   (HH * K1)   // gate-row stride in W1p, elements
#define G2   (HH * K2)   // gate-row stride in W2p, elements

// ---------------------------------------------------------------------------
// Pack [Whh | Wih] -> bf16 row-major, and sum biases. Grid-stride.
// ---------------------------------------------------------------------------
__global__ void pack_weights(const float* __restrict__ Wih1, const float* __restrict__ Whh1,
                             const float* __restrict__ bih1, const float* __restrict__ bhh1,
                             const float* __restrict__ Wih2, const float* __restrict__ Whh2,
                             const float* __restrict__ bih2, const float* __restrict__ bhh2,
                             __bf16* __restrict__ W1p, __bf16* __restrict__ W2p,
                             float* __restrict__ b1,  float* __restrict__ b2)
{
    const int idx    = blockIdx.x * blockDim.x + threadIdx.x;
    const int stride = gridDim.x * blockDim.x;

    for (int i = idx; i < 2048 * K1; i += stride) {
        int r = i / K1, k = i - r * K1;
        float v = (k < HH) ? Whh1[r * HH + k] : Wih1[r * IN + (k - HH)];
        W1p[i] = (__bf16)v;
    }
    for (int i = idx; i < 2048 * K2; i += stride) {
        int r = i >> 10, k = i & 1023;
        float v = (k < HH) ? Whh2[r * HH + k] : Wih2[r * HH + (k - HH)];
        W2p[i] = (__bf16)v;
    }
    for (int i = idx; i < 2048; i += stride) {
        b1[i] = bih1[i] + bhh1[i];
        b2[i] = bih2[i] + bhh2[i];
    }
}

// ---------------------------------------------------------------------------
// Fused 2-layer LSTM + FC. grid = 16 (batch tiles), block = 1024 (32 waves).
// ---------------------------------------------------------------------------
__global__ __launch_bounds__(1024)
void lstm2_fused(const float*  __restrict__ x,
                 const __bf16* __restrict__ W1p,
                 const __bf16* __restrict__ W2p,
                 const float*  __restrict__ b1,
                 const float*  __restrict__ b2,
                 const float*  __restrict__ Wfc,
                 const float*  __restrict__ bfc,
                 float*        __restrict__ out)
{
    __shared__ __bf16 sh1[16 * HH];   // layer-1 hidden state, bf16
    __shared__ __bf16 sh2[16 * HH];   // layer-2 hidden state, bf16
    __shared__ __bf16 sx [16 * IN];   // current x_t tile, bf16

    const int tid  = threadIdx.x;
    const int wave = tid >> 5;
    const int lane = tid & 31;
    const int b0   = blockIdx.x * 16;
    const int nrow = lane & 15;       // N index in tile; also A-matrix M row
    const int ksel = lane >> 4;       // half-wave selector for WMMA layout
    const int hcol = wave * 16 + nrow;

    // zero initial hidden state
    for (int i = tid; i < 16 * HH; i += 1024) {
        sh1[i] = (__bf16)0.0f;
        sh2[i] = (__bf16)0.0f;
    }

    float c1[8], c2[8];
#pragma unroll
    for (int j = 0; j < 8; ++j) { c1[j] = 0.0f; c2[j] = 0.0f; }

    float bias1[4], bias2[4];
#pragma unroll
    for (int g = 0; g < 4; ++g) {
        bias1[g] = b1[g * HH + hcol];
        bias2[g] = b2[g * HH + hcol];
    }

    // Single B-address base per layer: gate g at constant element offset g*G.
    const __bf16* w1lane = W1p + (size_t)hcol * K1 + ksel * 8;
    const __bf16* w2lane = W2p + (size_t)hcol * K2 + ksel * 8;

    // Fixed per-lane A-fragment bases.
    const __bf16* aH1 = &sh1[nrow * HH + ksel * 8];
    const __bf16* aH2 = &sh2[nrow * HH + ksel * 8];
    const __bf16* aX  = &sx [nrow * IN + ksel * 8];

    v8f acc0, acc1, acc2, acc3;
    __syncthreads();

    for (int t = 0; t < TT; ++t) {
        // stage x[:, t, :] -> LDS bf16 (16 rows x 64 cols, one elem/thread)
        {
            const int bl = tid >> 6;
            const int ii = tid & 63;
            sx[bl * IN + ii] = (__bf16)x[((size_t)(b0 + bl) * TT + t) * IN + ii];
        }
        __syncthreads();   // x_t + h2_{t-1} visible

        // ---------------- layer 1: gates = [h1 | x_t] @ W1p^T + b1 ----------
#pragma unroll
        for (int j = 0; j < 8; ++j) {
            acc0[j] = bias1[0]; acc1[j] = bias1[1];
            acc2[j] = bias1[2]; acc3[j] = bias1[3];
        }
        {
            const __bf16* pw = w1lane;
#pragma unroll 1
            for (int kt = 0; kt < 16; ++kt) {          // h1 segment of K
                v16bf bf0 = load_frag_32B(pw);
                v16bf bf1 = load_frag_32B(pw + 1 * G1);
                v16bf bf2 = load_frag_32B(pw + 2 * G1);
                v16bf bf3 = load_frag_32B(pw + 3 * G1);
                const __bf16* pa = aH1 + kt * 32;
                v16bf a = load_frag_2x16B(pa, pa + 16);
                acc0 = __builtin_amdgcn_wmma_f32_16x16x32_bf16(false, a, false, bf0, (short)0, acc0, false, false);
                acc1 = __builtin_amdgcn_wmma_f32_16x16x32_bf16(false, a, false, bf1, (short)0, acc1, false, false);
                acc2 = __builtin_amdgcn_wmma_f32_16x16x32_bf16(false, a, false, bf2, (short)0, acc2, false, false);
                acc3 = __builtin_amdgcn_wmma_f32_16x16x32_bf16(false, a, false, bf3, (short)0, acc3, false, false);
                pw += 32;
            }
#pragma unroll
            for (int kx = 0; kx < 2; ++kx) {           // x_t segment of K
                v16bf bf0 = load_frag_32B(pw);
                v16bf bf1 = load_frag_32B(pw + 1 * G1);
                v16bf bf2 = load_frag_32B(pw + 2 * G1);
                v16bf bf3 = load_frag_32B(pw + 3 * G1);
                const __bf16* pa = aX + kx * 32;
                v16bf a = load_frag_2x16B(pa, pa + 16);
                acc0 = __builtin_amdgcn_wmma_f32_16x16x32_bf16(false, a, false, bf0, (short)0, acc0, false, false);
                acc1 = __builtin_amdgcn_wmma_f32_16x16x32_bf16(false, a, false, bf1, (short)0, acc1, false, false);
                acc2 = __builtin_amdgcn_wmma_f32_16x16x32_bf16(false, a, false, bf2, (short)0, acc2, false, false);
                acc3 = __builtin_amdgcn_wmma_f32_16x16x32_bf16(false, a, false, bf3, (short)0, acc3, false, false);
                pw += 32;
            }
        }

        float h1v[8];
#pragma unroll
        for (int j = 0; j < 8; ++j) {
            float iv = sigm_f(acc0[j]);
            float fv = sigm_f(acc1[j]);
            float gv = tanh_f(acc2[j]);
            float ov = sigm_f(acc3[j]);
            c1[j]  = fv * c1[j] + iv * gv;
            h1v[j] = ov * tanh_f(c1[j]);
        }
        __syncthreads();   // all reads of h1_{t-1} complete
#pragma unroll
        for (int j = 0; j < 8; ++j) {
            const int m = j + 8 * ksel;               // D layout: M = j (+8 hi half)
            sh1[m * HH + hcol] = (__bf16)h1v[j];
        }
        __syncthreads();   // h1_t visible

        // ---------------- layer 2: gates = [h2 | h1_t] @ W2p^T + b2 ---------
#pragma unroll
        for (int j = 0; j < 8; ++j) {
            acc0[j] = bias2[0]; acc1[j] = bias2[1];
            acc2[j] = bias2[2]; acc3[j] = bias2[3];
        }
        {
            const __bf16* pw = w2lane;
#pragma unroll 1
            for (int kt = 0; kt < 16; ++kt) {          // h2 segment of K
                v16bf bf0 = load_frag_32B(pw);
                v16bf bf1 = load_frag_32B(pw + 1 * G2);
                v16bf bf2 = load_frag_32B(pw + 2 * G2);
                v16bf bf3 = load_frag_32B(pw + 3 * G2);
                const __bf16* pa = aH2 + kt * 32;
                v16bf a = load_frag_2x16B(pa, pa + 16);
                acc0 = __builtin_amdgcn_wmma_f32_16x16x32_bf16(false, a, false, bf0, (short)0, acc0, false, false);
                acc1 = __builtin_amdgcn_wmma_f32_16x16x32_bf16(false, a, false, bf1, (short)0, acc1, false, false);
                acc2 = __builtin_amdgcn_wmma_f32_16x16x32_bf16(false, a, false, bf2, (short)0, acc2, false, false);
                acc3 = __builtin_amdgcn_wmma_f32_16x16x32_bf16(false, a, false, bf3, (short)0, acc3, false, false);
                pw += 32;
            }
#pragma unroll 1
            for (int kt = 0; kt < 16; ++kt) {          // h1_t segment of K
                v16bf bf0 = load_frag_32B(pw);
                v16bf bf1 = load_frag_32B(pw + 1 * G2);
                v16bf bf2 = load_frag_32B(pw + 2 * G2);
                v16bf bf3 = load_frag_32B(pw + 3 * G2);
                const __bf16* pa = aH1 + kt * 32;
                v16bf a = load_frag_2x16B(pa, pa + 16);
                acc0 = __builtin_amdgcn_wmma_f32_16x16x32_bf16(false, a, false, bf0, (short)0, acc0, false, false);
                acc1 = __builtin_amdgcn_wmma_f32_16x16x32_bf16(false, a, false, bf1, (short)0, acc1, false, false);
                acc2 = __builtin_amdgcn_wmma_f32_16x16x32_bf16(false, a, false, bf2, (short)0, acc2, false, false);
                acc3 = __builtin_amdgcn_wmma_f32_16x16x32_bf16(false, a, false, bf3, (short)0, acc3, false, false);
                pw += 32;
            }
        }

        float h2v[8];
#pragma unroll
        for (int j = 0; j < 8; ++j) {
            float iv = sigm_f(acc0[j]);
            float fv = sigm_f(acc1[j]);
            float gv = tanh_f(acc2[j]);
            float ov = sigm_f(acc3[j]);
            c2[j]  = fv * c2[j] + iv * gv;
            h2v[j] = ov * tanh_f(c2[j]);
        }
        __syncthreads();   // all reads of h2_{t-1} complete
#pragma unroll
        for (int j = 0; j < 8; ++j) {
            const int m = j + 8 * ksel;
            sh2[m * HH + hcol] = (__bf16)h2v[j];
        }
        // next-iteration top barrier publishes h2_t before anyone reads it
    }

    __syncthreads();
    // FC: out[b] = h2_final[b, :] . Wfc + bfc   (O = 1, trivial)
    if (tid < 16) {
        float s = bfc[0];
        const __bf16* hp = &sh2[tid * HH];
        for (int k = 0; k < HH; ++k) s += (float)hp[k] * Wfc[k];
        out[b0 + tid] = s;
    }
}

// ---------------------------------------------------------------------------
// Host launcher
// ---------------------------------------------------------------------------
extern "C" void kernel_launch(void* const* d_in, const int* in_sizes, int n_in,
                              void* d_out, int out_size, void* d_ws, size_t ws_size,
                              hipStream_t stream)
{
    const float* x    = (const float*)d_in[0];
    const float* Wih1 = (const float*)d_in[1];
    const float* Whh1 = (const float*)d_in[2];
    const float* bih1 = (const float*)d_in[3];
    const float* bhh1 = (const float*)d_in[4];
    const float* Wih2 = (const float*)d_in[5];
    const float* Whh2 = (const float*)d_in[6];
    const float* bih2 = (const float*)d_in[7];
    const float* bhh2 = (const float*)d_in[8];
    const float* Wfc  = (const float*)d_in[9];
    const float* bfc  = (const float*)d_in[10];

    char* ws = (char*)d_ws;
    const size_t offW1 = 0;                          // 2048*576*2  = 2,359,296 B
    const size_t offW2 = offW1 + 2048 * 576 * 2;     // 2048*1024*2 = 4,194,304 B
    const size_t offB1 = offW2 + 2048 * 1024 * 2;    // 2048*4
    const size_t offB2 = offB1 + 2048 * 4;           // 2048*4

    __bf16* W1p = (__bf16*)(ws + offW1);
    __bf16* W2p = (__bf16*)(ws + offW2);
    float*  b1  = (float*)(ws + offB1);
    float*  b2  = (float*)(ws + offB2);

    pack_weights<<<1024, 256, 0, stream>>>(Wih1, Whh1, bih1, bhh1,
                                           Wih2, Whh2, bih2, bhh2,
                                           W1p, W2p, b1, b2);

    lstm2_fused<<<16, 1024, 0, stream>>>(x, W1p, W2p, b1, b2, Wfc, bfc,
                                         (float*)d_out);
}
